// ModelClass_91001767067935
// MI455X (gfx1250) — compile-verified, matrix operand order checked by reading, not run on hardware
//
#include <hip/hip_runtime.h>
#include <hip/hip_bf16.h>

// ---------------------------------------------------------------------------
// Hierarchical GNN forward (EPiC discriminator stack) for MI455X / gfx1250.
// One wave32 per graph; all hidden-40 FFNs run on v_wmma_f32_16x16x32_f16.
// All activation buffers live in LDS; buffers are selected by integer index
// (no pointer PHIs) and every helper is force-inlined so the compiler can
// infer addrspace(3) and emit ds_* ops instead of flat_*.
// LeakyReLU is expressed as max(v, 0.01*v) -> single v_max_num_f32.
// ---------------------------------------------------------------------------

typedef _Float16 v16h __attribute__((ext_vector_type(16)));
typedef _Float16 v8h  __attribute__((ext_vector_type(8)));
typedef float    v8f  __attribute__((ext_vector_type(8)));

#define STRIDE 64            // f16 elements per activation row (K padded to 64)
#define BUFELT (32 * STRIDE) // elements per 32x64 buffer
#define NW 4                 // waves (graphs) per workgroup

struct FfnW  { const float* w1; const float* w2; };
struct DiscW { FfnW disc, emb, glob, out; };
struct LevelW{ FfnW gin0, gin1, out; const float* pool; FfnW space; };
struct NetW  { DiscW discs[3]; DiscW last; LevelW levels[3]; };

// leaky_relu(v, 0.01) == max(v, 0.01*v): one v_mul + one v_max_num_f32.
__device__ __forceinline__ float leaky(float v) { return fmaxf(v, 0.01f * v); }

__device__ __forceinline__ v8f wmma32(v16h a, v16h b, v8f c) {
    return __builtin_amdgcn_wmma_f32_16x16x32_f16(false, a, false, b,
                                                  (short)0, c, false, false);
}

__device__ __forceinline__ _Float16* bufp(_Float16* sb, int idx) {
    return sb + idx * BUFELT;
}

// Zero one 32x64 f16 buffer (each lane clears its own row: 128 B).
__device__ __forceinline__ void zero_buf(_Float16* sb, int idx, int lane) {
    v8h z = {};
    v8h* p = (v8h*)(bufp(sb, idx) + lane * STRIDE);
#pragma unroll
    for (int j = 0; j < 8; ++j) p[j] = z;
}

// A fragment (16x32 f16) from LDS, row-major [node][feat], per ISA layout:
// lanes 0-15: M=lane, K kb..kb+7 (v0-3) and kb+16..kb+23 (v4-7); lanes 16-31: K+8.
__device__ __forceinline__ v16h lds_load_A(_Float16* sb, int idx, int mt, int kt, int lane) {
    int m  = (lane & 15) + mt * 16;
    int kb = ((lane >> 4) << 3) + kt * 32;
    const _Float16* p = bufp(sb, idx) + m * STRIDE + kb;
    v8h lo = *(const v8h*)(p);        // K kb .. kb+7
    v8h hi = *(const v8h*)(p + 16);   // K kb+16 .. kb+23
    v16h a;
#pragma unroll
    for (int j = 0; j < 8; ++j) { a[j] = lo[j]; a[8 + j] = hi[j]; }
    return a;
}

// B fragment (32x16, K x N) built from f32 global weights W[fin][fout].
// Same lane layout as A with N in place of M; zero-fills K/N padding.
__device__ __forceinline__ v16h load_B(const float* W, int fin, int fout,
                                       int kt, int nt, int lane) {
    int c  = (lane & 15) + nt * 16;
    int kb = ((lane >> 4) << 3) + kt * 32;
    v16h b = {};
    if (c < fout) {
#pragma unroll
        for (int half = 0; half < 2; ++half) {
            int k0 = kb + half * 16;
#pragma unroll
            for (int j = 0; j < 8; ++j) {
                int k = k0 + j;
                b[half * 8 + j] = (k < fin) ? (_Float16)W[k * fout + c] : (_Float16)0.f;
            }
        }
    }
    return b;
}

// Store C/D tile (f32) to LDS as f16, optional LeakyReLU.
// VGPR r: lanes 0-15 -> M=r, N=lane; lanes 16-31 -> M=r+8, N=lane-16.
__device__ __forceinline__ void store_D(_Float16* sb, int idx, v8f c, int mt, int nt,
                                        int lane, bool act) {
    _Float16* buf = bufp(sb, idx);
    int nn = (lane & 15) + nt * 16;
    int mb = ((lane >> 4) << 3) + mt * 16;
#pragma unroll
    for (int r = 0; r < 8; ++r) {
        float v = c[r];
        if (act) v = leaky(v);
        buf[(mb + r) * STRIDE + nn] = (_Float16)v;
    }
}

// Two-layer FFN  fin -> 40 (leaky) -> fout (leaky unless final_linear),
// entirely via WMMA.  fin <= 32 always; K2 = 40 padded to 64 (2 K-tiles).
// inb may equal outb (layer-1 consumes inb before outb is cleared).
__device__ __forceinline__ void wmma_ffn(_Float16* sb, int inb, int midb, int outb,
                                         const float* W1, const float* W2,
                                         int fin, int fout, int mtiles,
                                         bool final_linear, int lane) {
    __syncthreads();
    zero_buf(sb, midb, lane);
    __syncthreads();
    for (int mt = 0; mt < mtiles; ++mt) {
        v16h a = lds_load_A(sb, inb, mt, 0, lane);
#pragma unroll
        for (int nt = 0; nt < 3; ++nt) {        // 40 hidden -> 3 N-tiles
            v16h b = load_B(W1, fin, 40, 0, nt, lane);
            v8f c = {};
            c = wmma32(a, b, c);
            store_D(sb, midb, c, mt, nt, lane, true);
        }
    }
    __syncthreads();
    zero_buf(sb, outb, lane);
    __syncthreads();
    int n2 = (fout + 15) >> 4;
    for (int mt = 0; mt < mtiles; ++mt) {
        v16h a0 = lds_load_A(sb, midb, mt, 0, lane);
        v16h a1 = lds_load_A(sb, midb, mt, 1, lane);
        for (int nt = 0; nt < n2; ++nt) {
            v8f c = {};
            c = wmma32(a0, load_B(W2, 40, fout, 0, nt, lane), c);
            c = wmma32(a1, load_B(W2, 40, fout, 1, nt, lane), c);
            store_D(sb, outb, c, mt, nt, lane, !final_linear);
        }
    }
    __syncthreads();
}

// Per-graph vector FFN  5 -> 40 -> fout using VALU + LDS reduce.
// Every lane ends with the full vout (redundantly).
__device__ __forceinline__ void small_ffn(const float* vin5, int fout,
                                          const float* W1, const float* W2,
                                          bool final_linear, float* svec,
                                          float* vout, int lane) {
    int j1 = lane + 32;
    float h0 = 0.f, h1 = 0.f;
#pragma unroll
    for (int k = 0; k < 5; ++k) {
        float v = vin5[k];
        h0 += v * W1[k * 40 + lane];
        if (j1 < 40) h1 += v * W1[k * 40 + j1];
    }
    __syncthreads();
    svec[lane] = leaky(h0);
    if (j1 < 40) svec[j1] = leaky(h1);
    __syncthreads();
    for (int c = 0; c < fout; ++c) {
        float acc = 0.f;
        for (int j = 0; j < 40; ++j) acc += svec[j] * W2[j * fout + c];
        vout[c] = final_linear ? acc : leaky(acc);
    }
    __syncthreads();
}

// EPiC discriminator head: emb FFN -> pool -> glob FFN -> concat -> out FFN
// -> pool -> disc FFN(final linear).  Preserves buffer bA (the level input).
__device__ __forceinline__ float epic_disc(_Float16* sb, int bA, int b1, int b2,
                                           const DiscW& dw, int n, int fin,
                                           float* svec, int lane) {
    int mt = (n + 15) >> 4;
    wmma_ffn(sb, bA, b1, b2, dw.emb.w1, dw.emb.w2, fin, 5, mt, false, lane);  // h5 in b2
    _Float16* B2 = bufp(sb, b2);
    __syncthreads();
    if (lane < 5) {
        float s = 0.f;
        for (int i = 0; i < n; ++i) s += (float)B2[i * STRIDE + lane];
        svec[40 + lane] = s;
    }
    __syncthreads();
    float g5[5];
#pragma unroll
    for (int c = 0; c < 5; ++c) g5[c] = svec[40 + c];
    float g4[4];
    small_ffn(g5, 4, dw.glob.w1, dw.glob.w2, false, svec, g4, lane);
    zero_buf(sb, b1, lane);
    __syncthreads();
    _Float16* B1 = bufp(sb, b1);
    if (lane < n) {                               // concat [h5 | g4] per node
#pragma unroll
        for (int c = 0; c < 5; ++c) B1[lane * STRIDE + c] = B2[lane * STRIDE + c];
#pragma unroll
        for (int c = 0; c < 4; ++c) B1[lane * STRIDE + 5 + c] = (_Float16)g4[c];
    }
    __syncthreads();
    wmma_ffn(sb, b1, b2, b1, dw.out.w1, dw.out.w2, 9, 5, mt, false, lane);   // h5' in b1
    __syncthreads();
    if (lane < 5) {
        float s = 0.f;
        for (int i = 0; i < n; ++i) s += (float)B1[i * STRIDE + lane];
        svec[40 + lane] = s;
    }
    __syncthreads();
    float s5[5];
#pragma unroll
    for (int c = 0; c < 5; ++c) s5[c] = svec[40 + c];
    float o1[1];
    small_ffn(s5, 1, dw.disc.w1, dw.disc.w2, true, svec, o1, lane);
    return o1[0];
}

// level_layer + topk_pool.  On exit buffer index bA holds the pooled
// next-level features (indices rotated, not pointers).
__device__ __forceinline__ void level_layer(_Float16* sb, int& bA, int& b1, int& b2,
                                            const LevelW& lw, int n, int fin, int fout,
                                            int keff, int nnext,
                                            float* svec, unsigned char* sidx, int lane) {
    int mt = (n + 15) >> 4;
    wmma_ffn(sb, bA, b1, b2, lw.space.w1, lw.space.w2, fin, 10, mt, false, lane); // h in b2
    _Float16* B2 = bufp(sb, b2);
    float* dist = (float*)bufp(sb, bA);          // old x dead -> reuse as f32 dist
    __syncthreads();
    if (lane < n) {
        float px = (float)B2[lane * STRIDE + 0];
        float py = (float)B2[lane * STRIDE + 1];
        for (int j = 0; j < n; ++j) {
            float dx = px - (float)B2[j * STRIDE + 0];
            float dy = py - (float)B2[j * STRIDE + 1];
            float d = dx * dx + dy * dy;
            if (j == lane) d = 1e9f;
            dist[lane * 32 + j] = d;
        }
    }
    __syncthreads();
    if (lane < n) {                              // keff-NN selection per node
        for (int kk = 0; kk < keff; ++kk) {
            float best = 3.4e38f; int bj = 0;
            for (int j = 0; j < n; ++j) {
                float d = dist[lane * 32 + j];
                if (d < best) { best = d; bj = j; }
            }
            sidx[lane * 16 + kk] = (unsigned char)bj;
            dist[lane * 32 + bj] = 3.4e38f;
        }
    }
    __syncthreads();
    for (int gl = 0; gl < 2; ++gl) {             // 2 GIN layers with skip
        const FfnW& gw = gl ? lw.gin1 : lw.gin0;
        zero_buf(sb, b1, lane);
        __syncthreads();
        _Float16* B1 = bufp(sb, b1);
        if (lane < n) {
            float acc[10];
#pragma unroll
            for (int c = 0; c < 10; ++c) acc[c] = (float)B2[lane * STRIDE + c];
            for (int kk = 0; kk < keff; ++kk) {
                int j = sidx[lane * 16 + kk];
#pragma unroll
                for (int c = 0; c < 10; ++c) acc[c] += (float)B2[j * STRIDE + c];
            }
#pragma unroll
            for (int c = 0; c < 10; ++c) B1[lane * STRIDE + c] = (_Float16)acc[c];
        }
        __syncthreads();
        wmma_ffn(sb, b1, bA, b1, gw.w1, gw.w2, 10, 10, mt, false, lane);
        __syncthreads();
        if (lane < n) {                          // h = h + ffn(agg)
#pragma unroll
            for (int c = 0; c < 10; ++c) {
                float v = (float)B2[lane * STRIDE + c] + (float)B1[lane * STRIDE + c];
                B2[lane * STRIDE + c] = (_Float16)v;
            }
        }
        __syncthreads();
    }
    wmma_ffn(sb, b2, b1, bA, lw.out.w1, lw.out.w2, 10, fout, mt, false, lane);  // x_emb in bA
    _Float16* BA = bufp(sb, bA);
    __syncthreads();
    // TopK pooling: score = tanh(rowsum * sign(w))
    float w = *lw.pool;
    float sgn = (w >= 0.f) ? 1.f : -1.f;
    if (lane < n) {
        float s = 0.f;
        for (int c = 0; c < fout; ++c) s += (float)BA[lane * STRIDE + c];
        svec[lane] = tanhf(s * sgn);
    }
    __syncthreads();
    for (int r = 0; r < nnext; ++r) {            // argmax w/ low-index tie-break
        float best = -3.4e38f; int bj = 0;
        for (int j = 0; j < n; ++j) {
            float s = svec[j];
            if (s > best) { best = s; bj = j; }
        }
        if (lane == 0) { sidx[64 + r] = (unsigned char)bj; svec[bj] = -3.4e38f; }
        __syncthreads();
    }
    zero_buf(sb, b1, lane);
    __syncthreads();
    _Float16* B1 = bufp(sb, b1);
    if (lane < fout) {                           // gather pooled rows
        for (int r = 0; r < nnext; ++r)
            B1[r * STRIDE + lane] = BA[(int)sidx[64 + r] * STRIDE + lane];
    }
    __syncthreads();
    int t = bA; bA = b1; b1 = t;                 // pooled x becomes new bA
}

__global__ __launch_bounds__(128) void gnn_forward(NetW P, const float* __restrict__ x,
                                                   float* __restrict__ out, int G) {
    __shared__ __align__(16) _Float16 sbuf[NW][3][BUFELT];       // 48 KB
    __shared__ float         svec_s[NW][64];                     // 1 KB
    __shared__ unsigned char sidx_s[NW][512];                    // 2 KB

    const int tid  = threadIdx.x;
    const int wave = tid >> 5;
    const int lane = tid & 31;
    int g = blockIdx.x * NW + wave;
    if (g >= G) g = G - 1;                       // duplicate work, keeps barriers uniform

    _Float16* sb = &sbuf[wave][0][0];
    float* svec = svec_s[wave];
    unsigned char* sidx = sidx_s[wave];
    int bA = 0, b1 = 1, b2 = 2;

    const int nodes[4] = {30, 6, 2, 1};
    const int featsA[4] = {3, 6, 12, 18};
    const int keffA[3] = {15, 5, 1};

    zero_buf(sb, 0, lane); zero_buf(sb, 1, lane); zero_buf(sb, 2, lane);
    __syncthreads();
    if (lane < 30) {                             // load x[g] : 30 x 3 f32 -> f16
        const float* xr = x + ((long)g * 30 + lane) * 3;
        _Float16* BA = bufp(sb, 0);
#pragma unroll
        for (int c = 0; c < 3; ++c) BA[lane * STRIDE + c] = (_Float16)xr[c];
    }
    __syncthreads();

    float xd = 0.f;
    for (int L = 0; L < 3; ++L) {
        xd += epic_disc(sb, bA, b1, b2, P.discs[L], nodes[L], featsA[L], svec, lane);
        level_layer(sb, bA, b1, b2, P.levels[L], nodes[L], featsA[L], featsA[L + 1],
                    keffA[L], nodes[L + 1], svec, sidx, lane);
    }
    xd += epic_disc(sb, bA, b1, b2, P.last, 1, 18, svec, lane);
    if (lane == 0) out[g] = xd;
}

// ---------------------------------------------------------------------------
// Host-side marshalling.  Input order = jax pytree flatten of setup_inputs():
//   0: batch, 1: condition,
//   2..25 : discs[0..2] x {disc.w1,disc.w2, emb.w1,emb.w2, glob.w1,glob.w2, out.w1,out.w2}
//   26..33: last (same 8-leaf order)
//   34..60: levels[0..2] x {gin0.w1,gin0.w2, gin1.w1,gin1.w2, out.w1,out.w2, pool_w, space.w1,space.w2}
//   last  : x
// ---------------------------------------------------------------------------
extern "C" void kernel_launch(void* const* d_in, const int* in_sizes, int n_in,
                              void* d_out, int out_size, void* d_ws, size_t ws_size,
                              hipStream_t stream) {
    (void)in_sizes; (void)d_ws; (void)ws_size;
    auto F = [&](int i) { return (const float*)d_in[i]; };
    NetW P;
    int i = 2;
    for (int d = 0; d < 3; ++d) {
        P.discs[d].disc = {F(i + 0), F(i + 1)};
        P.discs[d].emb  = {F(i + 2), F(i + 3)};
        P.discs[d].glob = {F(i + 4), F(i + 5)};
        P.discs[d].out  = {F(i + 6), F(i + 7)};
        i += 8;
    }
    P.last.disc = {F(i + 0), F(i + 1)};
    P.last.emb  = {F(i + 2), F(i + 3)};
    P.last.glob = {F(i + 4), F(i + 5)};
    P.last.out  = {F(i + 6), F(i + 7)};
    i += 8;
    for (int l = 0; l < 3; ++l) {
        P.levels[l].gin0  = {F(i + 0), F(i + 1)};
        P.levels[l].gin1  = {F(i + 2), F(i + 3)};
        P.levels[l].out   = {F(i + 4), F(i + 5)};
        P.levels[l].pool  = F(i + 6);
        P.levels[l].space = {F(i + 7), F(i + 8)};
        i += 9;
    }
    const float* x = (const float*)d_in[n_in - 1];
    float* out = (float*)d_out;
    int G = out_size;                 // one output per graph
    int blocks = (G + NW - 1) / NW;   // one wave per graph, 4 waves per block
    gnn_forward<<<blocks, 128, 0, stream>>>(P, x, out, G);
}
